// SGC_76828374990959
// MI455X (gfx1250) — matrix operand order repaired and checked by reading, not compiled, not run on hardware
//
#include <hip/hip_runtime.h>
#include <hip/hip_bf16.h>
#include <cstdint>

typedef _Float16 f16;
typedef __attribute__((ext_vector_type(16))) _Float16 v16h;
typedef __attribute__((ext_vector_type(8)))  float    v8f;

#define B_   16
#define T_   12
#define N_   307
#define D_   64
#define C_VALID (T_*N_)   /* 3684 */
#define C_PAD   3712      /* multiple of 32 */
#define NPAD    320
#define NEG_INF_F (-1e30f)

// ---------------------------------------------------------------------------
// CDNA5 WMMA helper: D(16x16 f32) = A(16x32 f16) * B(32x16 f16) + C
// ---------------------------------------------------------------------------
__device__ __forceinline__ v8f wmma_f16f32(v16h a, v16h b, v8f c) {
  return __builtin_amdgcn_wmma_f32_16x16x32_f16(false, a, false, b, (short)0, c,
                                                false, false);
}

// A-operand per-lane gather for one 32-wide K chunk (row-major f16 source row).
// lanes 0-15 (lo=0): elems 0..15 = K0..15
// lanes 16-31(lo=8): elems 0..15 = K16..31
__device__ __forceinline__ v16h loadA_f16(const f16* row, int lo) {
  v16h a;
#pragma unroll
  for (int e = 0; e < 8; ++e) { a[e] = row[2 * lo + e]; a[e + 8] = row[2 * lo + 8 + e]; }
  return a;
}
__device__ __forceinline__ v16h loadA_f32(const float* row, int lo) {
  v16h a;
#pragma unroll
  for (int e = 0; e < 8; ++e) {
    a[e]     = (f16)row[2 * lo + e];
    a[e + 8] = (f16)row[2 * lo + 8 + e];
  }
  return a;
}
// B-operand: lane holds N=lane&15; K contiguous: lanes0-15 K=hk+0..15 (hk=0),
// lanes16-31 K=hk+0..15 (hk=16).  p points at first K element for this lane.
__device__ __forceinline__ v16h loadB_f16(const f16* p) {
  v16h b;
#pragma unroll
  for (int e = 0; e < 16; ++e) b[e] = p[e];
  return b;
}
__device__ __forceinline__ v16h loadB_f32(const float* p) {
  v16h b;
#pragma unroll
  for (int e = 0; e < 16; ++e) b[e] = (f16)p[e];
  return b;
}

// ---------------------------------------------------------------------------
// K1: LayerNorm(x) -> x_f16 (padded [b][C_PAD][64]) and y_f32 (t=11 slice, f32)
// One wave per row, 2 elements per lane, wave32 shuffle reductions.
// ---------------------------------------------------------------------------
__global__ __launch_bounds__(256) void ln_kernel(
    const float* __restrict__ x, const float* __restrict__ g,
    const float* __restrict__ be, f16* __restrict__ xf, float* __restrict__ yf) {
  const int lane = threadIdx.x & 31;
  int wid = (blockIdx.x * blockDim.x + threadIdx.x) >> 5;
  const int nw = (gridDim.x * blockDim.x) >> 5;
  const float g0 = g[lane], g1 = g[lane + 32];
  const float e0 = be[lane], e1 = be[lane + 32];
  for (int pr = wid; pr < B_ * C_PAD; pr += nw) {
    const int b = pr / C_PAD, c = pr - b * C_PAD;
    f16* orow = xf + (size_t)pr * D_;
    if (c >= C_VALID) { orow[lane] = (f16)0.f; orow[lane + 32] = (f16)0.f; continue; }
    const float* row = x + ((size_t)b * C_VALID + c) * D_;
    const float v0 = row[lane], v1 = row[lane + 32];
    float s = v0 + v1;
#pragma unroll
    for (int m = 16; m >= 1; m >>= 1) s += __shfl_xor(s, m, 32);
    const float mu = s * (1.f / 64.f);
    const float d0 = v0 - mu, d1 = v1 - mu;
    float q = d0 * d0 + d1 * d1;
#pragma unroll
    for (int m = 16; m >= 1; m >>= 1) q += __shfl_xor(q, m, 32);
    const float rs = rsqrtf(q * (1.f / 64.f) + 1e-5f);
    const float o0 = d0 * rs * g0 + e0, o1 = d1 * rs * g1 + e1;
    orow[lane] = (f16)o0; orow[lane + 32] = (f16)o1;
    if (c >= (T_ - 1) * N_) {  // y_ = x_[:, -1]
      float* yr = yf + ((size_t)b * N_ + (c - (T_ - 1) * N_)) * D_;
      yr[lane] = o0; yr[lane + 32] = o1;
    }
  }
}

// ---------------------------------------------------------------------------
// K2a: k = x_ @ k_w^T + k_b  ->  k_f16 [b][C_PAD][64].  One wave per 16-row tile.
// ---------------------------------------------------------------------------
__global__ __launch_bounds__(256) void kproj_kernel(
    const f16* __restrict__ xf, const float* __restrict__ kw,
    const float* __restrict__ kbias, f16* __restrict__ kf) {
  const int lane = threadIdx.x & 31;
  const int nn = lane & 15;
  const int lo = (lane < 16) ? 0 : 8;
  const int hk = (lane < 16) ? 0 : 16;
  const int mr = (lane < 16) ? 0 : 8;
  int wid = (blockIdx.x * blockDim.x + threadIdx.x) >> 5;
  const int nw = (gridDim.x * blockDim.x) >> 5;
  const int NT = C_PAD / 16;  // 232 tiles per batch
  for (int tile = wid; tile < B_ * NT; tile += nw) {
    const int b = tile / NT, c0 = (tile - b * NT) * 16;
    const f16* arow = xf + ((size_t)b * C_PAD + c0 + nn) * D_;
    const v16h a0 = loadA_f16(arow, lo), a1 = loadA_f16(arow + 32, lo);
    for (int d4 = 0; d4 < 4; ++d4) {
      const int dcol = d4 * 16 + nn;
      const float* wr = kw + (size_t)dcol * D_;  // B[kk][nn] = kw[dcol][kk]
      v8f acc = {};
      acc = wmma_f16f32(a0, loadB_f32(wr + hk), acc);
      acc = wmma_f16f32(a1, loadB_f32(wr + 32 + hk), acc);
      const float bs = kbias[dcol];
#pragma unroll
      for (int v = 0; v < 8; ++v)
        kf[((size_t)b * C_PAD + c0 + mr + v) * D_ + dcol] = (f16)(acc[v] + bs);
    }
  }
}

// ---------------------------------------------------------------------------
// K2b: q = y_ @ q_w^T + q_b  ->  q_f16 [b][NPAD][64]
// ---------------------------------------------------------------------------
__global__ __launch_bounds__(256) void qproj_kernel(
    const f16* __restrict__ xf, const float* __restrict__ qw,
    const float* __restrict__ qbias, f16* __restrict__ qf) {
  const int lane = threadIdx.x & 31;
  const int nn = lane & 15;
  const int lo = (lane < 16) ? 0 : 8;
  const int hk = (lane < 16) ? 0 : 16;
  const int mr = (lane < 16) ? 0 : 8;
  int wid = (blockIdx.x * blockDim.x + threadIdx.x) >> 5;
  const int nw = (gridDim.x * blockDim.x) >> 5;
  const int NT = NPAD / 16;  // 20
  for (int tile = wid; tile < B_ * NT; tile += nw) {
    const int b = tile / NT, n0 = (tile - b * NT) * 16;
    const int n = n0 + nn;
    v16h a0, a1;
    if (n < N_) {  // y_ row = x_f16[b][(T_-1)*N_ + n]
      const f16* arow = xf + ((size_t)b * C_PAD + (T_ - 1) * N_ + n) * D_;
      a0 = loadA_f16(arow, lo); a1 = loadA_f16(arow + 32, lo);
    } else {
#pragma unroll
      for (int e = 0; e < 16; ++e) { a0[e] = (f16)0.f; a1[e] = (f16)0.f; }
    }
    for (int d4 = 0; d4 < 4; ++d4) {
      const int dcol = d4 * 16 + nn;
      const float* wr = qw + (size_t)dcol * D_;
      v8f acc = {};
      acc = wmma_f16f32(a0, loadB_f32(wr + hk), acc);
      acc = wmma_f16f32(a1, loadB_f32(wr + 32 + hk), acc);
      const float bs = qbias[dcol];
#pragma unroll
      for (int v = 0; v < 8; ++v)
        qf[((size_t)b * NPAD + n0 + mr + v) * D_ + dcol] = (f16)(acc[v] + bs);
    }
  }
}

// ---------------------------------------------------------------------------
// K3: attention core. 1 workgroup = (batch b, 16-row n-tile); 8 waves.
// LDS: sag[16][C_PAD] f32 (238KB) + hist[8][256] + opart[8][16][64]  ~272KB
// Phase1: dg = q k^T via WMMA, gate = sigmoid(dg*s)*stg*s -> LDS
// Phase2: exact kth-largest per row (4-pass radix select), filter, softmax
// Phase3: o = sag @ x_ via WMMA (K split across waves), h = o + y_
// ---------------------------------------------------------------------------
__global__ __launch_bounds__(256) void attn_kernel(
    const float* __restrict__ stg, const int* __restrict__ topk_p,
    const f16* __restrict__ qf, const f16* __restrict__ kf,
    const f16* __restrict__ xf, const float* __restrict__ yf,
    float* __restrict__ hf) {
  extern __shared__ char smem[];
  float*    sag   = (float*)smem;                                      // 16*C_PAD
  unsigned* hist  = (unsigned*)(smem + (size_t)16 * C_PAD * 4);        // 8*256
  float*    opart = (float*)(smem + (size_t)16 * C_PAD * 4 + 8 * 256 * 4);

  const int nt = blockIdx.x, b = blockIdx.y;
  const int n0 = nt * 16;
  const int lane = threadIdx.x & 31, wv = threadIdx.x >> 5;
  const int nn = lane & 15;
  const int lo = (lane < 16) ? 0 : 8;
  const int hk = (lane < 16) ? 0 : 16;
  const int mr = (lane < 16) ? 0 : 8;
  const float scale = 0.125f;  // 1/sqrt(64)

  // ---- Phase 1: scores ----
  const f16* qrow = qf + ((size_t)b * NPAD + n0 + nn) * D_;
  const v16h a0 = loadA_f16(qrow, lo), a1 = loadA_f16(qrow + 32, lo);
  for (int j = wv; j < C_PAD / 16; j += 8) {   // 232/8 = 29: uniform
    const int c0 = j * 16;
    const f16* krow = kf + ((size_t)b * C_PAD + c0 + nn) * D_;
    v8f acc = {};
    acc = wmma_f16f32(a0, loadB_f16(krow + hk), acc);
    acc = wmma_f16f32(a1, loadB_f16(krow + 32 + hk), acc);
    const int cc = c0 + nn;
    if (cc < C_VALID) {
      const int t = cc / N_, m = cc - t * N_;
#pragma unroll
      for (int v = 0; v < 8; ++v) {
        int n = n0 + mr + v; if (n > N_ - 1) n = N_ - 1;  // clamp OOB rows
        const float sg = 1.f / (1.f + __expf(-acc[v] * scale));
        const float val =
            sg * stg[(((size_t)b * N_ + n) * T_ + t) * N_ + m] * scale;
        sag[(mr + v) * C_PAD + cc] = val;
      }
    } else {
#pragma unroll
      for (int v = 0; v < 8; ++v) sag[(mr + v) * C_PAD + cc] = NEG_INF_F;
    }
  }
  __syncthreads();

  // ---- Phase 2: top-k threshold + softmax; wave wv owns rows wv, wv+8 ----
  const int tk = topk_p[0];
  const int krank = (tk > 0) ? ((tk < 5) ? tk * N_ : tk) : 0;  // block-uniform
  for (int rr = 0; rr < 2; ++rr) {
    const int r = wv + rr * 8;
    float thr = -3.0e38f;
    if (krank > 0) {
      unsigned prefix = 0; int kr = krank;
      for (int pass = 0; pass < 4; ++pass) {
        const int shift = 24 - 8 * pass;
        for (int i = lane; i < 256; i += 32) hist[wv * 256 + i] = 0u;
        __syncthreads();
        for (int c = lane; c < C_PAD; c += 32) {
          unsigned u = __float_as_uint(sag[r * C_PAD + c]);
          u = (u & 0x80000000u) ? ~u : (u | 0x80000000u);  // order-preserving map
          const bool sel = (pass == 0) || ((u >> (shift + 8)) == prefix);
          if (sel) atomicAdd(&hist[wv * 256 + ((u >> shift) & 255u)], 1u);
        }
        __syncthreads();
        unsigned cum = 0, dsel = 0;
        for (int dgt = 255; dgt >= 0; --dgt) {       // redundant per-lane scan
          const unsigned h = hist[wv * 256 + dgt];
          if (cum + h >= (unsigned)kr) { dsel = (unsigned)dgt; kr -= (int)cum; break; }
          cum += h;
        }
        prefix = (prefix << 8) | dsel;
        __syncthreads();
      }
      thr = __uint_as_float((prefix & 0x80000000u) ? (prefix ^ 0x80000000u)
                                                   : ~prefix);
    }
    // softmax over filtered row (strict > thr keeps < k entries, as reference)
    float mx = -3.0e38f;
    for (int c = lane; c < C_PAD; c += 32) {
      float f = sag[r * C_PAD + c];
      f = (f > thr) ? f : NEG_INF_F;
      sag[r * C_PAD + c] = f;
      mx = fmaxf(mx, f);
    }
#pragma unroll
    for (int m = 16; m >= 1; m >>= 1) mx = fmaxf(mx, __shfl_xor(mx, m, 32));
    float sum = 0.f;
    for (int c = lane; c < C_PAD; c += 32) {
      const float e = __expf(sag[r * C_PAD + c] - mx);
      sag[r * C_PAD + c] = e;
      sum += e;
    }
#pragma unroll
    for (int m = 16; m >= 1; m >>= 1) sum += __shfl_xor(sum, m, 32);
    const float inv = 1.f / sum;
    for (int c = lane; c < C_PAD; c += 32) sag[r * C_PAD + c] *= inv;
  }
  __syncthreads();

  // ---- Phase 3: o = sag @ x_ ; waves split the 116 K-tiles ----
  v8f zero = {};
  v8f oacc[4] = {zero, zero, zero, zero};
  for (int kt = wv; kt < C_PAD / 32; kt += 8) {
    const int kb = kt * 32;
    const v16h aa = loadA_f32(sag + nn * C_PAD + kb, lo);
    const f16* xbase = xf + ((size_t)b * C_PAD + kb + hk) * D_;
#pragma unroll
    for (int d4 = 0; d4 < 4; ++d4) {
      const f16* xp = xbase + d4 * 16 + nn;
      v16h bb;
#pragma unroll
      for (int e = 0; e < 16; ++e) bb[e] = xp[(size_t)e * D_];
      oacc[d4] = wmma_f16f32(aa, bb, oacc[d4]);
    }
  }
#pragma unroll
  for (int d4 = 0; d4 < 4; ++d4)
#pragma unroll
    for (int v = 0; v < 8; ++v)
      opart[(wv * 16 + mr + v) * D_ + d4 * 16 + nn] = oacc[d4][v];
  __syncthreads();
  for (int idx = threadIdx.x; idx < 16 * D_; idx += 256) {
    float s = 0.f;
#pragma unroll
    for (int w = 0; w < 8; ++w) s += opart[w * 16 * D_ + idx];
    const int M = idx >> 6, d = idx & 63;
    const int n = n0 + M;
    if (n < N_) {
      const size_t off = ((size_t)b * N_ + n) * D_ + d;
      hf[off] = s + yf[off];  // h = o + y_
    }
  }
}

// ---------------------------------------------------------------------------
// K4: FFN. One wave per 16-row tile (4912 rows = 307 tiles exactly).
// z = LN(h); out = h + relu(z W1^T + b1) W2^T + b2.  Per-wave LDS staging.
// ---------------------------------------------------------------------------
__global__ __launch_bounds__(256) void ffn_kernel(
    const float* __restrict__ hf, const float* __restrict__ g,
    const float* __restrict__ be, const float* __restrict__ w1,
    const float* __restrict__ b1, const float* __restrict__ w2,
    const float* __restrict__ b2, float* __restrict__ out) {
  extern __shared__ char smem[];
  const int lane = threadIdx.x & 31, wv = threadIdx.x >> 5;
  f16* zbuf = (f16*)(smem + (size_t)wv * (16 * 64 + 16 * 256) * 2);
  f16* rbuf = zbuf + 16 * 64;
  const int tile = blockIdx.x * 8 + wv;
  if (tile >= (B_ * N_) / 16) return;  // 307 tiles; wave-uniform, no block barriers
  const int r0 = tile * 16;
  const int nn = lane & 15;
  const int lo = (lane < 16) ? 0 : 8;
  const int hk = (lane < 16) ? 0 : 16;
  const int mr = (lane < 16) ? 0 : 8;

  if (lane < 16) {  // LayerNorm, one row per lane
    const float* hr = hf + (size_t)(r0 + lane) * D_;
    float s = 0.f;
    for (int k = 0; k < D_; ++k) s += hr[k];
    const float mu = s * (1.f / 64.f);
    float q = 0.f;
    for (int k = 0; k < D_; ++k) { const float dd = hr[k] - mu; q += dd * dd; }
    const float rs = rsqrtf(q * (1.f / 64.f) + 1e-5f);
    for (int k = 0; k < D_; ++k)
      zbuf[lane * D_ + k] = (f16)((hr[k] - mu) * rs * g[k] + be[k]);
  }
  asm volatile("s_wait_dscnt 0" ::: "memory");
  __builtin_amdgcn_wave_barrier();

  const f16* zr = zbuf + nn * D_;
  const v16h a0 = loadA_f16(zr, lo), a1 = loadA_f16(zr + 32, lo);
  for (int j = 0; j < 16; ++j) {  // hidden 256 in 16 N-tiles
    const int i = j * 16 + nn;
    const float* wr = w1 + (size_t)i * D_;
    v8f acc = {};
    acc = wmma_f16f32(a0, loadB_f32(wr + hk), acc);
    acc = wmma_f16f32(a1, loadB_f32(wr + 32 + hk), acc);
    const float bs = b1[i];
#pragma unroll
    for (int v = 0; v < 8; ++v) {
      float val = acc[v] + bs;
      val = val > 0.f ? val : 0.f;  // relu
      rbuf[(mr + v) * 256 + i] = (f16)val;
    }
  }
  asm volatile("s_wait_dscnt 0" ::: "memory");
  __builtin_amdgcn_wave_barrier();

  const f16* rr0 = rbuf + nn * 256;
  for (int d4 = 0; d4 < 4; ++d4) {
    const int dcol = d4 * 16 + nn;
    const float* wr = w2 + (size_t)dcol * 256;
    v8f acc = {};
    for (int kt = 0; kt < 8; ++kt) {  // K = 256
      const int kb = kt * 32;
      acc = wmma_f16f32(loadA_f16(rr0 + kb, lo), loadB_f32(wr + kb + hk), acc);
    }
    const float bs = b2[dcol];
#pragma unroll
    for (int v = 0; v < 8; ++v) {
      const size_t off = (size_t)(r0 + mr + v) * D_ + dcol;
      out[off] = hf[off] + acc[v] + bs;  // h + z
    }
  }
}

// ---------------------------------------------------------------------------
extern "C" void kernel_launch(void* const* d_in, const int* in_sizes, int n_in,
                              void* d_out, int out_size, void* d_ws,
                              size_t ws_size, hipStream_t stream) {
  (void)in_sizes; (void)n_in; (void)out_size; (void)ws_size;
  const float* x    = (const float*)d_in[0];
  const float* stg  = (const float*)d_in[1];
  const int*   topk = (const int*)d_in[2];
  const float* q_w  = (const float*)d_in[3];
  const float* q_b  = (const float*)d_in[4];
  const float* k_w  = (const float*)d_in[5];
  const float* k_b  = (const float*)d_in[6];
  const float* ln_g = (const float*)d_in[7];
  const float* ln_b = (const float*)d_in[8];
  const float* fg   = (const float*)d_in[9];
  const float* fb   = (const float*)d_in[10];
  const float* w1   = (const float*)d_in[11];
  const float* b1   = (const float*)d_in[12];
  const float* w2   = (const float*)d_in[13];
  const float* b2   = (const float*)d_in[14];
  float* out = (float*)d_out;

  char* ws = (char*)d_ws;
  size_t off = 0;
  auto carve = [&](size_t bytes) -> char* {
    char* p = ws + off;
    off += (bytes + 255) & ~(size_t)255;
    return p;
  };
  f16*   xf = (f16*)carve((size_t)B_ * C_PAD * D_ * 2);   // LN(x), f16, padded
  f16*   kf = (f16*)carve((size_t)B_ * C_PAD * D_ * 2);   // k proj, f16
  f16*   qf = (f16*)carve((size_t)B_ * NPAD * D_ * 2);    // q proj, f16
  float* yf = (float*)carve((size_t)B_ * N_ * D_ * 4);    // y_ (f32)
  float* hf = (float*)carve((size_t)B_ * N_ * D_ * 4);    // h (f32)

  const size_t attn_lds =
      (size_t)16 * C_PAD * 4 + 8 * 256 * 4 + 8 * 16 * 64 * 4;  // ~272 KB < 320 KB WGP LDS
  const size_t ffn_lds = (size_t)8 * (16 * 64 + 16 * 256) * 2; // 80 KB
  (void)hipFuncSetAttribute((const void*)attn_kernel,
                            hipFuncAttributeMaxDynamicSharedMemorySize,
                            (int)attn_lds);
  (void)hipFuncSetAttribute((const void*)ffn_kernel,
                            hipFuncAttributeMaxDynamicSharedMemorySize,
                            (int)ffn_lds);

  ln_kernel<<<464, 256, 0, stream>>>(x, ln_g, ln_b, xf, yf);
  kproj_kernel<<<464, 256, 0, stream>>>(xf, k_w, k_b, kf);
  qproj_kernel<<<40, 256, 0, stream>>>(xf, q_w, q_b, qf);
  attn_kernel<<<dim3(20, 16), 256, attn_lds, stream>>>(stg, topk, qf, kf, xf,
                                                       yf, hf);
  ffn_kernel<<<39, 256, ffn_lds, stream>>>(hf, fg, fb, w1, b1, w2, b2, out);
}
